// DifferentiableMPO_70497593197463
// MI455X (gfx1250) — matrix operand order repaired and checked by reading, not compiled, not run on hardware
//
#include <hip/hip_runtime.h>
#include <hip/hip_bf16.h>

#define GAMMA_RA     5.0f
#define COST_COEFF   1e-3f
#define KAPPA        2.0627128f
#define CVAR_PEN     50.0f
#define SP_BETA      50.0f
#define MD_ETA       0.05f
#define MAX_ITERS    300
#define NB           128      // B
#define NH           12       // H
#define NN           100      // N
#define NPAD         112      // padded row stride for M (7*16)

typedef __attribute__((ext_vector_type(16))) _Float16 v16h;
typedef __attribute__((ext_vector_type(8)))  float    v8f;
typedef __attribute__((ext_vector_type(4)))  _Float16 h4;

#if defined(__AMDGCN__) && __has_builtin(__builtin_amdgcn_tensor_load_to_lds)
#define HAVE_TDM 1
typedef unsigned int u32x4 __attribute__((ext_vector_type(4)));
typedef int          i32x4 __attribute__((ext_vector_type(4)));
typedef int          i32x8 __attribute__((ext_vector_type(8)));
#endif

// ---------------------------------------------------------------------------
// Phase 1: M[bh] = L[bh] * L[bh]^T  (batched SYRK via v_wmma_f32_16x16x32_f16)
// grid = (B*H, 7, 7) tiles of 16x16; one wave32 per tile; K padded 100 -> 128.
// Output M stored as f16 (inputs to the WMMA were f16 already).
// ---------------------------------------------------------------------------
__global__ __launch_bounds__(32)
void syrk_llt_kernel(const float* __restrict__ L, _Float16* __restrict__ M) {
    const int bh   = blockIdx.x;          // 0..1535
    const int ti   = blockIdx.y;          // 0..6 (row tile)
    const int tj   = blockIdx.z;          // 0..6 (col tile)
    const int lane = threadIdx.x;         // 0..31 (wave32)

    const float*  Lb = L + (size_t)bh * NN * NN;
    _Float16*     Mb = M + (size_t)bh * NN * NPAD;

    // Stage the two 16-row blocks of L into LDS, coalesced.
    __shared__ float As[16 * NN];
    __shared__ float Bs[16 * NN];
    for (int idx = lane; idx < 16 * NN; idx += 32) {
        const int r = idx / NN, c = idx - r * NN;
        const int gi = ti * 16 + r;
        const int gj = tj * 16 + r;
        As[idx] = (gi < NN) ? Lb[gi * NN + c] : 0.0f;
        Bs[idx] = (gj < NN) ? Lb[gj * NN + c] : 0.0f;
    }
    __syncthreads();

    v8f acc = {0.f, 0.f, 0.f, 0.f, 0.f, 0.f, 0.f, 0.f};
    const int mrow = lane & 15;           // A: M row / B: N col
    const int hi   = (lane & 16) ? 1 : 0; // lane half

#pragma unroll
    for (int k0 = 0; k0 < 128; k0 += 32) {
        v16h a, b;
#pragma unroll
        for (int e = 0; e < 16; ++e) {
            // A-fragment (16-bit 16x32): VGPR r holds K pairs; lane halves
            // split K as 0-7/16-23 (lanes 0-15) vs 8-15/24-31 (lanes 16-31).
            const int ra = e >> 1;
            const int ka = k0 + ((ra >> 2) << 4) + hi * 8 + ((ra & 3) << 1) + (e & 1);
            a[e] = (ka < NN) ? (_Float16)As[mrow * NN + ka] : (_Float16)0.0f;
            // B-fragment (16-bit 32x16): same K split as A: lanes 0-15 hold
            // K 0-7/16-23, lanes 16-31 hold K 8-15/24-31.  B[k][n] = L[j0+n][k].
            const int kb = k0 + ((e >> 3) << 4) + hi * 8 + (e & 7);
            b[e] = (kb < NN) ? (_Float16)Bs[mrow * NN + kb] : (_Float16)0.0f;
        }
        acc = __builtin_amdgcn_wmma_f32_16x16x32_f16(
            /*neg_a=*/false, a, /*neg_b=*/false, b,
            /*c_mod=*/(short)0, acc, /*reuse_a=*/false, /*reuse_b=*/false);
    }

    // C/D f32 16x16: VGPR v -> M row v + 8*half, lane%16 -> N col.
    const int col = tj * 16 + mrow;
#pragma unroll
    for (int v = 0; v < 8; ++v) {
        const int row = ti * 16 + v + hi * 8;
        if (row < NN) Mb[row * NPAD + col] = (_Float16)acc[v];
    }
}

// ---------------------------------------------------------------------------
// Phase 2: mirror-descent fixed-point loop. One block per batch b, one wave32
// per horizon h.  All 12 M_h slices (f16, 268.8 KB) are staged into the WGP's
// 320 KB LDS once (TDM bulk DMA + coalesced backstop copy); the 300
// iterations then run entirely out of LDS.
// Lane owns 4 consecutive assets (N = 100 = 4*25 -> lanes 0..24 active).
// ---------------------------------------------------------------------------
__global__ __launch_bounds__(NH * 32)
void md_iter_kernel(const float* __restrict__ mu, const _Float16* __restrict__ M,
                    const float* __restrict__ w_prev,
                    const float* __restrict__ cvar_limit,
                    float* __restrict__ out) {
    const int b    = blockIdx.x;
    const int h    = threadIdx.x >> 5;
    const int lane = threadIdx.x & 31;
    const int tid  = threadIdx.x;

    extern __shared__ char smem[];
    _Float16* M_lds = (_Float16*)smem;                       // NH*NN*NPAD f16
    float*    w_sh  = (float*)(smem + NH * NN * NPAD * 2);   // NH*NPAD f32
    float*    wp_sh = w_sh + NH * NPAD;                      // NPAD f32

#ifdef HAVE_TDM
    // Tensor Data Mover: DMA M[b] (268,800 B) global -> LDS offset 0 as a
    // 1400x24 tile of 8-byte units.  Single-issue from wave 0; TENSORcnt
    // tracks completion.  D# per ISA ch.8 tables.
    if (h == 0) {
        const unsigned long long ga =
            (unsigned long long)(const void*)(M + (size_t)b * (NH * NN * NPAD));
        u32x4 g0;
        g0[0] = 1u;                                    // count=1, user mode
        g0[1] = 0u;                                    // lds_addr = 0 (M_lds base)
        g0[2] = (unsigned)(ga & 0xFFFFFFFFull);        // global_addr[31:0]
        g0[3] = (unsigned)((ga >> 32) & 0x1FFFFFFull)  // global_addr[56:32]
              | (2u << 30);                            // type = 2 ("image")
        i32x8 g1;
        g1[0] = (3 << 16);                 // data_size=3 (8B), no multicast
        g1[1] = (1400 & 0xFFFF) << 16;     // tensor_dim0[15:0] = 1400
        g1[2] = (24 << 16);                // tensor_dim0[31:16]=0 | tensor_dim1=24
        g1[3] = (1400 << 16);              // tensor_dim1 hi=0 | tile_dim0=1400
        g1[4] = 24;                        // tile_dim1=24, tile_dim2=0
        g1[5] = 1400;                      // tensor_dim0_stride[31:0]
        g1[6] = 0;                         // stride hi / dim1_stride lo
        g1[7] = 0;
        i32x4 gz = {0, 0, 0, 0};
#if __clang_major__ >= 23
        i32x8 gz8 = {0, 0, 0, 0, 0, 0, 0, 0};
        __builtin_amdgcn_tensor_load_to_lds(g0, g1, gz, gz, gz8, 0);
#else
        __builtin_amdgcn_tensor_load_to_lds(g0, g1, gz, gz, 0);
#endif
        __builtin_amdgcn_s_wait_tensorcnt(0);
    }
    __syncthreads();   // TDM data visible to whole workgroup
#endif

    // Coalesced copy of M[b] -> LDS (backstop for / alongside the TDM path;
    // reads hit L2 that the DMA just warmed).
    {
        const uint4* src = (const uint4*)(M + (size_t)b * NH * NN * NPAD);
        uint4*       dst = (uint4*)M_lds;
        const int nvec = NH * NN * NPAD / 8;   // 8 f16 per uint4
        for (int idx = tid; idx < nvec; idx += NH * 32) dst[idx] = src[idx];
    }
    for (int idx = tid; idx < NH * NPAD; idx += NH * 32) {
        const int c = idx % NPAD;
        w_sh[idx] = (c < NN) ? (1.0f / NN) : 0.0f;
    }
    for (int idx = tid; idx < NPAD; idx += NH * 32)
        wp_sh[idx] = (idx < NN) ? w_prev[b * NN + idx] : 0.0f;
    __syncthreads();

    const float     cvar   = cvar_limit[b];
    const _Float16* Mh_lds = M_lds + h * NN * NPAD;
    const float*    muh    = mu + (size_t)(b * NH + h) * NN;
    float*          wh     = w_sh + h * NPAD;

    const int  i0 = 4 * lane;          // 4 consecutive assets per lane
    const bool ok = (i0 < NN);         // lanes 0..24 active
    float mur[4];
#pragma unroll
    for (int r = 0; r < 4; ++r) mur[r] = ok ? muh[i0 + r] : 0.0f;

    for (int it = 0; it < MAX_ITERS; ++it) {
        float wv[4], q[4];
#pragma unroll
        for (int r = 0; r < 4; ++r) {
            wv[r] = ok ? wh[i0 + r] : 0.0f;
            q[r]  = 0.0f;
        }

        // q = M w ; per j: one ds_load_b64 of 4 f16 per lane (conflict-free),
        // w_j broadcast from LDS.  M symmetric so column == row access.
#pragma unroll 4
        for (int j = 0; j < NN; ++j) {
            const float wj = wh[j];
            if (ok) {
                const h4 m = *(const h4*)(Mh_lds + j * NPAD + i0);
#pragma unroll
                for (int r = 0; r < 4; ++r)
                    q[r] = fmaf((float)m[r], wj, q[r]);
            }
        }

        // wave reductions: mu_p = mu.w ; sigma^2 = w.q
        float pmu = 0.0f, pwq = 0.0f;
#pragma unroll
        for (int r = 0; r < 4; ++r) { pmu += mur[r] * wv[r]; pwq += wv[r] * q[r]; }
#pragma unroll
        for (int m = 16; m >= 1; m >>= 1) {
            pmu += __shfl_xor(pmu, m, 32);
            pwq += __shfl_xor(pwq, m, 32);
        }
        const float sigma = sqrtf(pwq);
        const float isig  = 1.0f / sigma;
        const float viol  = -pmu + KAPPA * sigma - cvar;
        const float smul  = CVAR_PEN / (1.0f + __expf(-SP_BETA * viol)); // pen*sigmoid

        // gradient + MD update + wave softmax over N=100
        float logw[4];
        float lmax = -__builtin_inff();
#pragma unroll
        for (int r = 0; r < 4; ++r) {
            if (!ok) { logw[r] = -__builtin_inff(); continue; }
            const float wi = wv[r];
            const float wm = (h == 0) ? wp_sh[i0 + r] : w_sh[(h - 1) * NPAD + i0 + r];
            const float d0 = wi - wm;
            const float c0 = d0 / sqrtf(d0 * d0 + 1e-8f);
            float c1 = 0.0f;
            if (h < NH - 1) {
                const float d1 = w_sh[(h + 1) * NPAD + i0 + r] - wi;
                c1 = d1 / sqrtf(d1 * d1 + 1e-8f);
            }
            const float g = -mur[r] + 2.0f * GAMMA_RA * q[r]
                          + COST_COEFF * (c0 - c1)
                          + smul * (-mur[r] + KAPPA * q[r] * isig);
            logw[r] = __logf(wi + 1e-10f) - MD_ETA * g;
            lmax = fmaxf(lmax, logw[r]);
        }
#pragma unroll
        for (int m = 16; m >= 1; m >>= 1)
            lmax = fmaxf(lmax, __shfl_xor(lmax, m, 32));

        float ev[4], psum = 0.0f;
#pragma unroll
        for (int r = 0; r < 4; ++r) {
            ev[r] = ok ? __expf(logw[r] - lmax) : 0.0f;
            psum += ev[r];
        }
#pragma unroll
        for (int m = 16; m >= 1; m >>= 1) psum += __shfl_xor(psum, m, 32);
        const float inv_sum = 1.0f / psum;

        __syncthreads();   // neighbors done reading old w
        if (ok) {
#pragma unroll
            for (int r = 0; r < 4; ++r) wh[i0 + r] = ev[r] * inv_sum;
        }
        __syncthreads();   // new w visible to neighbors
    }

    if (ok) {
#pragma unroll
        for (int r = 0; r < 4; ++r)
            out[(size_t)(b * NH + h) * NN + i0 + r] = wh[i0 + r];
    }
}

// ---------------------------------------------------------------------------
extern "C" void kernel_launch(void* const* d_in, const int* in_sizes, int n_in,
                              void* d_out, int out_size, void* d_ws, size_t ws_size,
                              hipStream_t stream) {
    const float* mu     = (const float*)d_in[0];  // [B,H,N]
    const float* L      = (const float*)d_in[1];  // [B,H,N,N]
    const float* w_prev = (const float*)d_in[2];  // [B,N]
    const float* cvar   = (const float*)d_in[3];  // [B]
    _Float16* M = (_Float16*)d_ws;                // [B*H, 100, 112] f16 = 34.4 MB

    dim3 g1(NB * NH, 7, 7);
    syrk_llt_kernel<<<g1, 32, 0, stream>>>(L, M);

    const size_t smem = (size_t)NH * NN * NPAD * 2   // M in f16
                      + (size_t)NH * NPAD * 4        // w iterate
                      + (size_t)NPAD * 4;            // w_prev
    md_iter_kernel<<<NB, NH * 32, smem, stream>>>(mu, M, w_prev, cvar, (float*)d_out);
}
